// BaseEncoderRNN_89000312308183
// MI455X (gfx1250) — compile-verified, matrix operand order checked by reading, not run on hardware
//
#include <hip/hip_runtime.h>
#include <hip/hip_bf16.h>
#include <stdint.h>

// GRU: H = hidden = input = 2048, L = seq len = 2048, batch 1.
#define H   2048
#define L   2048
#define K3  (3 * H)   // 6144

typedef __attribute__((ext_vector_type(16))) __bf16 v16bf;
typedef __attribute__((ext_vector_type(8)))  float  v8f;

union Frag16 {
    uint4  q[2];          // two 16-byte halves
    v16bf  v;             // WMMA fragment view
};

__device__ __forceinline__ unsigned short f32_to_bf16(float f) {
    unsigned u = __float_as_uint(f);
    u += 0x7FFFu + ((u >> 16) & 1u);       // round-to-nearest-even
    return (unsigned short)(u >> 16);
}
__device__ __forceinline__ float bf16_to_f32(unsigned short s) {
    return __uint_as_float(((unsigned)s) << 16);
}

// One 16-byte async global->LDS copy (per active lane). ASYNCcnt-tracked.
__device__ __forceinline__ void async_cp16(unsigned lds_off, const void* gptr) {
    asm volatile("global_load_async_to_lds_b128 %0, %1, off"
                 :: "v"(lds_off), "v"(gptr) : "memory");
}
__device__ __forceinline__ unsigned lds_off_of(const void* p) {
    // generic LDS address carries the wave-relative LDS offset in addr[31:0]
    return (unsigned)(unsigned long long)p;
}

// ---------------------------------------------------------------- converters
__global__ void cvt_f32_bf16(const float* __restrict__ src,
                             unsigned short* __restrict__ dst, int n) {
    int i = blockIdx.x * blockDim.x + threadIdx.x;
    int stride = gridDim.x * blockDim.x;
    for (; i < n; i += stride) dst[i] = f32_to_bf16(src[i]);
}

__global__ void init_state(float* __restrict__ h_buf, unsigned* __restrict__ bar) {
    for (int i = threadIdx.x; i < 2 * H; i += blockDim.x) h_buf[i] = 0.0f;
    if (threadIdx.x == 0) { bar[0] = 0u; bar[1] = 0u; }
}

// ------------------------------------------------- phase 1: Gi = X*W_ih^T + b
// X: (L,H) bf16 row-major.  W: (3H,H) bf16 row-major (N-major == B fragment).
// 8 waves / WG arranged 2x4; each wave -> 32x32 C tile; WG tile 64x128.
// K stepped by 32; tiles staged in LDS via async global->LDS, double-buffered.
__global__ void __launch_bounds__(256)
gemm_gi(const unsigned short* __restrict__ Xb,
        const unsigned short* __restrict__ Wb,
        const float* __restrict__ b_ih,
        float* __restrict__ Gi) {
    __shared__ __align__(16) uint4 Abuf[2][64 * 4];    // 64 rows x 32 bf16
    __shared__ __align__(16) uint4 Bbuf[2][128 * 4];   // 128 rows x 32 bf16

    const int tid  = threadIdx.x;
    const int lane = tid & 31;
    const int wave = tid >> 5;
    const int wm   = wave >> 2;              // 0..1
    const int wn   = wave & 3;               // 0..3
    const int half = lane >> 4;              // 0/1 : which K-half this lane holds
    const int l16  = lane & 15;              // row (A) / col (B) within 16
    const int m0   = blockIdx.y * 64;
    const int n0   = blockIdx.x * 128;

    // Per-thread staging slots: one 16B chunk of A, two 16B chunks of B.
    const int arow = tid >> 2, achk = tid & 3;              // A: 64 rows x 4 chunks
    const int brow0 = tid >> 2;                             // B rows 0..63
    const int brow1 = (tid >> 2) + 64;                      // B rows 64..127
    const int bchk  = tid & 3;
    const char* agp  = (const char*)Xb + ((size_t)(m0 + arow ) * H + achk * 8) * 2;
    const char* bgp0 = (const char*)Wb + ((size_t)(n0 + brow0) * H + bchk * 8) * 2;
    const char* bgp1 = (const char*)Wb + ((size_t)(n0 + brow1) * H + bchk * 8) * 2;

    auto issue = [&](int buf, int kstep) {
        const size_t koff = (size_t)kstep * 64;             // 32 elems * 2B
        async_cp16(lds_off_of(&Abuf[buf][arow  * 4 + achk]), agp  + koff);
        async_cp16(lds_off_of(&Bbuf[buf][brow0 * 4 + bchk]), bgp0 + koff);
        async_cp16(lds_off_of(&Bbuf[buf][brow1 * 4 + bchk]), bgp1 + koff);
    };

    v8f acc[2][2] = {};
    issue(0, 0);

    const int KSTEPS = H / 32;
    for (int s = 0; s < KSTEPS; ++s) {
        const int buf = s & 1;
        if (s + 1 < KSTEPS) {
            issue(buf ^ 1, s + 1);   // prefetch next tile into other buffer
            // async loads complete in order: <=3 outstanding => previous group done
            asm volatile("s_wait_asynccnt 0x3" ::: "memory");
        } else {
            asm volatile("s_wait_asynccnt 0x0" ::: "memory");
        }
        __syncthreads();             // all threads' tiles landed in LDS

        Frag16 a[2], b[2];
#pragma unroll
        for (int i = 0; i < 2; ++i) {
            // 16-bit A layout: lanes 0-15 hold K[0..7]+[16..23], lanes 16-31 the rest
            const int ar = wm * 32 + i * 16 + l16;
            a[i].q[0] = Abuf[buf][ar * 4 + half];
            a[i].q[1] = Abuf[buf][ar * 4 + 2 + half];
            const int br = wn * 32 + i * 16 + l16;
            b[i].q[0] = Bbuf[buf][br * 4 + half];
            b[i].q[1] = Bbuf[buf][br * 4 + 2 + half];
        }
#pragma unroll
        for (int i = 0; i < 2; ++i)
#pragma unroll
            for (int j = 0; j < 2; ++j)
                acc[i][j] = __builtin_amdgcn_wmma_f32_16x16x32_bf16(
                    /*neg_a=*/false, a[i].v, /*neg_b=*/false, b[j].v,
                    /*c_mod=*/(short)0, acc[i][j],
                    /*reuse_a=*/false, /*reuse_b=*/false);
        __syncthreads();             // done reading buf before it is overwritten
    }

    // Epilogue: C layout -> VGPR r holds M = r (lanes 0-15) / 8+r (lanes 16-31).
#pragma unroll
    for (int i = 0; i < 2; ++i) {
#pragma unroll
        for (int j = 0; j < 2; ++j) {
            const int   n    = n0 + wn * 32 + j * 16 + l16;
            const float bias = b_ih[n];
#pragma unroll
            for (int r = 0; r < 8; ++r) {
                const int m = m0 + wm * 32 + i * 16 + half * 8 + r;
                Gi[(size_t)m * K3 + n] = acc[i][j][r] + bias;
            }
        }
    }
}

// --------------------------------------------- phase 2: sequential recurrence
__device__ __forceinline__ void grid_sync(unsigned* bar, unsigned nblocks) {
    __syncthreads();
    if (threadIdx.x == 0) {
        __threadfence();  // make this block's h writes device-visible
        unsigned gen = __hip_atomic_load(&bar[1], __ATOMIC_RELAXED,
                                         __HIP_MEMORY_SCOPE_AGENT);
        unsigned arrived = __hip_atomic_fetch_add(&bar[0], 1u, __ATOMIC_ACQ_REL,
                                                  __HIP_MEMORY_SCOPE_AGENT);
        if (arrived == nblocks - 1u) {
            __hip_atomic_store(&bar[0], 0u, __ATOMIC_RELAXED,
                               __HIP_MEMORY_SCOPE_AGENT);
            __hip_atomic_fetch_add(&bar[1], 1u, __ATOMIC_RELEASE,
                                   __HIP_MEMORY_SCOPE_AGENT);
        } else {
            while (__hip_atomic_load(&bar[1], __ATOMIC_ACQUIRE,
                                     __HIP_MEMORY_SCOPE_AGENT) == gen)
                __builtin_amdgcn_s_sleep(1);
        }
    }
    __syncthreads();
}

// 256 blocks x 256 threads = 2048 waves: wave w owns hidden unit j = w.
// W_hh (3H x H bf16, 24 MB) stays L2-resident across all 2048 steps.
__global__ void __launch_bounds__(256)
gru_recur(const unsigned short* __restrict__ Whh,
          const float* __restrict__ Gi,
          const float* __restrict__ b_hh,
          float* __restrict__ h_buf,          // 2*H double buffer
          float* __restrict__ out,            // L*H outputs + 2*H hidden copies
          unsigned* __restrict__ bar) {
    const int lane = threadIdx.x & 31;
    const int j    = blockIdx.x * 8 + (threadIdx.x >> 5);   // 0..H-1
    const unsigned nblocks = gridDim.x;

    const unsigned short* w_r = Whh + (size_t)j * H;
    const unsigned short* w_z = Whh + (size_t)(j + H) * H;
    const unsigned short* w_n = Whh + (size_t)(j + 2 * H) * H;

    for (int t = 0; t < L; ++t) {
        const float* h  = h_buf + (t & 1) * H;
        float*       hn = h_buf + ((t + 1) & 1) * H;

        float s_r = 0.0f, s_z = 0.0f, s_n = 0.0f;
#pragma unroll 4
        for (int k = lane; k < H; k += 32) {
            const float hk = h[k];
            s_r = fmaf(bf16_to_f32(w_r[k]), hk, s_r);
            s_z = fmaf(bf16_to_f32(w_z[k]), hk, s_z);
            s_n = fmaf(bf16_to_f32(w_n[k]), hk, s_n);
        }
#pragma unroll
        for (int off = 16; off; off >>= 1) {   // wave32 reduction
            s_r += __shfl_xor(s_r, off, 32);
            s_z += __shfl_xor(s_z, off, 32);
            s_n += __shfl_xor(s_n, off, 32);
        }

        if (lane == 0) {
            const float* gi = Gi + (size_t)t * K3;
            const float hr = s_r + b_hh[j];
            const float hz = s_z + b_hh[j + H];
            const float hc = s_n + b_hh[j + 2 * H];
            const float r  = 1.0f / (1.0f + __expf(-(gi[j] + hr)));
            const float z  = 1.0f / (1.0f + __expf(-(gi[j + H] + hz)));
            const float n  = tanhf(gi[j + 2 * H] + r * hc);
            const float hnew = (1.0f - z) * n + z * h[j];
            hn[j] = hnew;
            out[(size_t)t * H + j] = hnew;
            if (t == L - 1) {                       // hidden returned twice
                out[(size_t)L * H + j]     = hnew;
                out[(size_t)L * H + H + j] = hnew;
            }
        }
        grid_sync(bar, nblocks);
    }
}

// --------------------------------------------------------------------- launch
extern "C" void kernel_launch(void* const* d_in, const int* in_sizes, int n_in,
                              void* d_out, int out_size, void* d_ws, size_t ws_size,
                              hipStream_t stream) {
    const float* x    = (const float*)d_in[0];
    const float* w_ih = (const float*)d_in[1];
    const float* w_hh = (const float*)d_in[2];
    const float* b_ih = (const float*)d_in[3];
    const float* b_hh = (const float*)d_in[4];
    float* out = (float*)d_out;

    char*  ws  = (char*)d_ws;
    size_t off = 0;
    unsigned short* x_bf   = (unsigned short*)(ws + off); off += (size_t)L  * H * 2;
    unsigned short* wih_bf = (unsigned short*)(ws + off); off += (size_t)K3 * H * 2;
    unsigned short* whh_bf = (unsigned short*)(ws + off); off += (size_t)K3 * H * 2;
    off = (off + 255) & ~(size_t)255;
    float*    Gi    = (float*)(ws + off); off += (size_t)L * K3 * 4;
    float*    h_buf = (float*)(ws + off); off += (size_t)2 * H * 4;
    unsigned* bar   = (unsigned*)(ws + off); off += 256;

    cvt_f32_bf16<<<512, 256, 0, stream>>>(x,    x_bf,   L * H);
    cvt_f32_bf16<<<512, 256, 0, stream>>>(w_ih, wih_bf, K3 * H);
    cvt_f32_bf16<<<512, 256, 0, stream>>>(w_hh, whh_bf, K3 * H);
    init_state<<<1, 256, 0, stream>>>(h_buf, bar);

    dim3 g(K3 / 128, L / 64);                  // 48 x 32 workgroups
    gemm_gi<<<g, 256, 0, stream>>>(x_bf, wih_bf, b_ih, Gi);

    gru_recur<<<256, 256, 0, stream>>>(whh_bf, Gi, b_hh, h_buf, out, bar);
}
